// Attention_27109833572866
// MI455X (gfx1250) — compile-verified, matrix operand order checked by reading, not run on hardware
//
#include <hip/hip_runtime.h>
#include <hip/hip_bf16.h>

// ---------------------------------------------------------------------------
// Dual-key attention for MI455X (gfx1250, wave32, WMMA bf16 16x16x32,
// async global->LDS staging).  B=4, C=256, CQ=64, N=64*64=4096.
// ---------------------------------------------------------------------------

#define BB   4
#define CC   256
#define CQ   64
#define NTOK 4096

typedef __attribute__((ext_vector_type(16))) __bf16 v16bf;
typedef __attribute__((ext_vector_type(8)))  float  v8f;

union BF16Frag { v16bf v; unsigned int u[8]; };
union F32Acc   { v8f   v; float f[8]; };

__device__ __forceinline__ unsigned short f2bf(float x) {
    unsigned int u = __float_as_uint(x);
    u += 0x7FFFu + ((u >> 16) & 1u);           // round-to-nearest-even
    return (unsigned short)(u >> 16);
}
__device__ __forceinline__ unsigned int pack2bf(float lo, float hi) {
    return (unsigned int)f2bf(lo) | ((unsigned int)f2bf(hi) << 16);
}
__device__ __forceinline__ unsigned int ld_u32(const unsigned short* p) {
    return *(const unsigned int*)p;
}
__device__ __forceinline__ v8f wmma_bf16(const BF16Frag& a, const BF16Frag& b, const F32Acc& c) {
    return __builtin_amdgcn_wmma_f32_16x16x32_bf16(
        false, a.v, false, b.v, (short)0, c.v, false, false);
}

// gfx1250 async copy: global memory -> LDS, 16 bytes per active lane.
__device__ __forceinline__ void async_ld_b128(unsigned lds_off, const void* gaddr) {
    asm volatile("global_load_async_to_lds_b128 %0, %1, off"
                 :: "v"(lds_off), "v"(gaddr) : "memory");
}
__device__ __forceinline__ void wait_asynccnt0() {
    asm volatile("s_wait_asynccnt 0x0" ::: "memory");
}

// ---------------------------------------------------------------------------
// Kernel 1: 1x1 conv projection  Y = W(OxC) * X(B,C,N) + b  -> bf16 workspace.
// TRANSPOSED=1: layout (B,N,O); TRANSPOSED=0: layout (B,O,N).
// ---------------------------------------------------------------------------
template <int TRANSPOSED>
__global__ __launch_bounds__(256)
void proj_kernel(const float* __restrict__ W, const float* __restrict__ bias,
                 const float* __restrict__ X, unsigned short* __restrict__ Y,
                 int O, int Cdim)
{
    const int lane = threadIdx.x & 31;
    const int wave = threadIdx.x >> 5;
    const int half = lane >> 4;
    const int l    = lane & 15;

    const int tiles_n = NTOK / 16;
    const int tiles_o = O / 16;
    const int tile = blockIdx.x * 8 + wave;
    const int b   = tile / (tiles_o * tiles_n);
    const int rem = tile % (tiles_o * tiles_n);
    const int o0  = (rem / tiles_n) * 16;
    const int n0  = (rem % tiles_n) * 16;

    const float* Xb = X + (size_t)b * Cdim * NTOK;

    F32Acc acc;
#pragma unroll
    for (int i = 0; i < 8; ++i) acc.f[i] = 0.0f;

    for (int k0 = 0; k0 < Cdim; k0 += 32) {
        BF16Frag A, Bf;
        const float* wrow = W + (size_t)(o0 + l) * Cdim + k0 + half * 8;
#pragma unroll
        for (int j = 0; j < 4; ++j) {
            A.u[j]     = pack2bf(wrow[2 * j],      wrow[2 * j + 1]);
            A.u[j + 4] = pack2bf(wrow[16 + 2 * j], wrow[16 + 2 * j + 1]);
        }
        const float* xcol = Xb + (size_t)(k0 + half * 8) * NTOK + (n0 + l);
#pragma unroll
        for (int j = 0; j < 4; ++j) {
            Bf.u[j]     = pack2bf(xcol[(size_t)(2 * j) * NTOK],      xcol[(size_t)(2 * j + 1) * NTOK]);
            Bf.u[j + 4] = pack2bf(xcol[(size_t)(16 + 2 * j) * NTOK], xcol[(size_t)(17 + 2 * j) * NTOK]);
        }
        acc.v = wmma_bf16(A, Bf, acc);
    }

#pragma unroll
    for (int r = 0; r < 8; ++r) {
        const int orow = o0 + r + 8 * half;
        const int ncol = n0 + l;
        const unsigned short bf = f2bf(acc.f[r] + bias[orow]);
        if (TRANSPOSED) Y[(size_t)b * NTOK * O + (size_t)ncol * O + orow] = bf;
        else            Y[(size_t)b * O * NTOK + (size_t)orow * NTOK + ncol] = bf;
    }
}

// ---------------------------------------------------------------------------
// Dual QK 16x16 logit tiles, interleaved accumulator chains so each bf16
// WMMA has independent work in its hazard window (no v_nop padding):
//   l1[n,m] = sum_c qT[n,c]*k1T[m,c],  l2[n,m] = sum_c qT[n,c]*k2T[m,c]
// ---------------------------------------------------------------------------
__device__ __forceinline__ void qk_tile2(const BF16Frag& A0, const BF16Frag& A1,
                                         const unsigned short* __restrict__ k1b,
                                         const unsigned short* __restrict__ k2b,
                                         int m0, int half, int l,
                                         F32Acc& acc1, F32Acc& acc2)
{
    BF16Frag B10, B11, B20, B21;
    const unsigned short* k1r = k1b + (size_t)(m0 + l) * CQ;
    const unsigned short* k2r = k2b + (size_t)(m0 + l) * CQ;
    const int c0 = half * 8;
#pragma unroll
    for (int j = 0; j < 4; ++j) {
        B10.u[j]     = ld_u32(k1r + c0 + 2 * j);
        B10.u[j + 4] = ld_u32(k1r + c0 + 16 + 2 * j);
        B11.u[j]     = ld_u32(k1r + 32 + c0 + 2 * j);
        B11.u[j + 4] = ld_u32(k1r + 32 + c0 + 16 + 2 * j);
        B20.u[j]     = ld_u32(k2r + c0 + 2 * j);
        B20.u[j + 4] = ld_u32(k2r + c0 + 16 + 2 * j);
        B21.u[j]     = ld_u32(k2r + 32 + c0 + 2 * j);
        B21.u[j + 4] = ld_u32(k2r + 32 + c0 + 16 + 2 * j);
    }
    acc1.v = wmma_bf16(A0, B10, acc1);
    acc2.v = wmma_bf16(A0, B20, acc2);   // independent: fills acc1's hazard slots
    acc1.v = wmma_bf16(A1, B11, acc1);
    acc2.v = wmma_bf16(A1, B21, acc2);
}

// ---------------------------------------------------------------------------
// Kernel 2: per 16-row block: online softmax stats for l1 and l2 (pass A),
// then recompute tiles, write exp(softmax(l1)+softmax(l2)) (pass B).
// ---------------------------------------------------------------------------
__global__ __launch_bounds__(256)
void attn_kernel(const unsigned short* __restrict__ qT,
                 const unsigned short* __restrict__ k1T,
                 const unsigned short* __restrict__ k2T,
                 float* __restrict__ attn,      // (B,N,N) in d_out, unnormalized exp
                 float* __restrict__ rowsum)    // (B,N)
{
    __shared__ float red[8][16][4];
    __shared__ float fin[16][4];
    __shared__ float redE[8][16];

    const int tid  = threadIdx.x;
    const int wave = tid >> 5;
    const int lane = tid & 31;
    const int half = lane >> 4;
    const int l    = lane & 15;

    const int b  = blockIdx.x / (NTOK / 16);
    const int n0 = (blockIdx.x % (NTOK / 16)) * 16;

    const unsigned short* qb  = qT  + (size_t)b * NTOK * CQ;
    const unsigned short* k1b = k1T + (size_t)b * NTOK * CQ;
    const unsigned short* k2b = k2T + (size_t)b * NTOK * CQ;

    BF16Frag A0, A1;
    {
        const unsigned short* qrow = qb + (size_t)(n0 + l) * CQ;
        const int c0 = half * 8;
#pragma unroll
        for (int j = 0; j < 4; ++j) {
            A0.u[j]     = ld_u32(qrow + c0 + 2 * j);
            A0.u[j + 4] = ld_u32(qrow + c0 + 16 + 2 * j);
            A1.u[j]     = ld_u32(qrow + 32 + c0 + 2 * j);
            A1.u[j + 4] = ld_u32(qrow + 32 + c0 + 16 + 2 * j);
        }
    }

    // ---- Pass A: lane-local online (max, sum) for l1 and l2 ----
    float rmax1[8], rsum1[8], rmax2[8], rsum2[8];
#pragma unroll
    for (int r = 0; r < 8; ++r) { rmax1[r] = rmax2[r] = -1e30f; rsum1[r] = rsum2[r] = 0.0f; }

    for (int mt = wave; mt < NTOK / 16; mt += 8) {
        F32Acc acc1, acc2;
#pragma unroll
        for (int i = 0; i < 8; ++i) { acc1.f[i] = 0.0f; acc2.f[i] = 0.0f; }
        qk_tile2(A0, A1, k1b, k2b, mt * 16, half, l, acc1, acc2);
#pragma unroll
        for (int r = 0; r < 8; ++r) {
            float x = acc1.f[r];
            float mn = fmaxf(rmax1[r], x);
            rsum1[r] = rsum1[r] * __expf(rmax1[r] - mn) + __expf(x - mn);
            rmax1[r] = mn;
            x  = acc2.f[r];
            mn = fmaxf(rmax2[r], x);
            rsum2[r] = rsum2[r] * __expf(rmax2[r] - mn) + __expf(x - mn);
            rmax2[r] = mn;
        }
    }
#pragma unroll
    for (int r = 0; r < 8; ++r) {
        for (int off = 8; off >= 1; off >>= 1) {
            float om = __shfl_xor(rmax1[r], off);
            float os = __shfl_xor(rsum1[r], off);
            float mn = fmaxf(rmax1[r], om);
            rsum1[r] = rsum1[r] * __expf(rmax1[r] - mn) + os * __expf(om - mn);
            rmax1[r] = mn;
            om = __shfl_xor(rmax2[r], off);
            os = __shfl_xor(rsum2[r], off);
            mn = fmaxf(rmax2[r], om);
            rsum2[r] = rsum2[r] * __expf(rmax2[r] - mn) + os * __expf(om - mn);
            rmax2[r] = mn;
        }
        if (l == 0) {
            const int row = r + 8 * half;
            red[wave][row][0] = rmax1[r]; red[wave][row][1] = rsum1[r];
            red[wave][row][2] = rmax2[r]; red[wave][row][3] = rsum2[r];
        }
    }
    __syncthreads();
    if (tid < 16) {
        float m1 = -1e30f, s1 = 0.0f, m2 = -1e30f, s2 = 0.0f;
#pragma unroll
        for (int w = 0; w < 8; ++w) {
            float om = red[w][tid][0], os = red[w][tid][1];
            float mn = fmaxf(m1, om);
            s1 = s1 * __expf(m1 - mn) + os * __expf(om - mn); m1 = mn;
            om = red[w][tid][2]; os = red[w][tid][3];
            mn = fmaxf(m2, om);
            s2 = s2 * __expf(m2 - mn) + os * __expf(om - mn); m2 = mn;
        }
        fin[tid][0] = m1; fin[tid][1] = 1.0f / s1;
        fin[tid][2] = m2; fin[tid][3] = 1.0f / s2;
    }
    __syncthreads();

    float fm1[8], fi1[8], fm2[8], fi2[8], esum[8];
#pragma unroll
    for (int r = 0; r < 8; ++r) {
        const int row = r + 8 * half;
        fm1[r] = fin[row][0]; fi1[r] = fin[row][1];
        fm2[r] = fin[row][2]; fi2[r] = fin[row][3];
        esum[r] = 0.0f;
    }

    // ---- Pass B: recompute tiles, stream out unnormalized exp(a1+a2) ----
    // a1+a2 in [0,2] so exp never overflows; no max subtraction needed here.
    float* ab = attn + (size_t)b * NTOK * NTOK;
    for (int mt = wave; mt < NTOK / 16; mt += 8) {
        const int m0 = mt * 16;
        F32Acc acc1, acc2;
#pragma unroll
        for (int i = 0; i < 8; ++i) { acc1.f[i] = 0.0f; acc2.f[i] = 0.0f; }
        qk_tile2(A0, A1, k1b, k2b, m0, half, l, acc1, acc2);
#pragma unroll
        for (int r = 0; r < 8; ++r) {
            const float a1 = __expf(acc1.f[r] - fm1[r]) * fi1[r];
            const float a2 = __expf(acc2.f[r] - fm2[r]) * fi2[r];
            const float e  = __expf(a1 + a2);
            const int row  = r + 8 * half;
            __builtin_nontemporal_store(e, &ab[(size_t)(n0 + row) * NTOK + m0 + l]);
            esum[r] += e;
        }
    }
#pragma unroll
    for (int r = 0; r < 8; ++r) {
        for (int off = 8; off >= 1; off >>= 1) esum[r] += __shfl_xor(esum[r], off);
        if (l == 0) redE[wave][r + 8 * half] = esum[r];
    }
    __syncthreads();
    if (tid < 16) {
        float t = 0.0f;
#pragma unroll
        for (int w = 0; w < 8; ++w) t += redE[w][tid];
        rowsum[(size_t)b * NTOK + n0 + tid] = t;
    }
}

// ---------------------------------------------------------------------------
// Kernel 3: normalize attn rows in place (final attn output).
// ---------------------------------------------------------------------------
__global__ __launch_bounds__(256)
void scale_kernel(float* __restrict__ attn, const float* __restrict__ rowsum)
{
    const int row = blockIdx.x;                  // b*N + n
    const float inv = 1.0f / rowsum[row];
    float4* p = (float4*)(attn + (size_t)row * NTOK);
    for (int i = threadIdx.x; i < NTOK / 4; i += 256) {
        float4 v = p[i];
        v.x *= inv; v.y *= inv; v.z *= inv; v.w *= inv;
        p[i] = v;
    }
}

// ---------------------------------------------------------------------------
// Kernel 4: out[b,c,n] = gamma * sum_m V[b,c,m]*attn[b,n,m] + Xres[b,c,n]
// One workgroup per (b, 16-wide n-tile) computes ALL 256 c's (8 waves x
// 2 c-tiles).  The shared attn 16n x 32m block is staged global->LDS with
// async b128 copies, double-buffered; attn HBM traffic = the 268 MB minimum.
// ---------------------------------------------------------------------------
__global__ __launch_bounds__(256)
void av_kernel(const unsigned short* __restrict__ V,    // (B,C,N) bf16
               const float* __restrict__ attn,          // (B,N,N) normalized
               const float* __restrict__ Xres,          // (B,C,N)
               const float* __restrict__ gamma,
               float* __restrict__ Out)                 // (B,C,N)
{
    __shared__ float atile[2][16 * 32];                 // double-buffered 16n x 32m

    const int tid  = threadIdx.x;
    const int wave = tid >> 5;
    const int lane = tid & 31;
    const int half = lane >> 4;
    const int l    = lane & 15;

    const int b  = blockIdx.x / (NTOK / 16);
    const int n0 = (blockIdx.x % (NTOK / 16)) * 16;
    const int c0 = wave * 32;                           // this wave: c0 and c0+16

    const unsigned short* Vb = V    + (size_t)b * CC * NTOK;
    const float*          Ab = attn + (size_t)b * NTOK * NTOK;
    const unsigned lds_base = (unsigned)(size_t)(void*)&atile[0][0];

    F32Acc acc0, acc1;
#pragma unroll
    for (int i = 0; i < 8; ++i) { acc0.f[i] = 0.0f; acc1.f[i] = 0.0f; }

    const int srow = tid >> 3;                          // 0..15
    const int smo  = (tid & 7) * 4;                     // 0,4,..,28

#define AV_STAGE(m0_, buf_)                                                        \
    if (tid < 128) {                                                               \
        async_ld_b128(lds_base + (unsigned)(((buf_) * 512 + srow * 32 + smo) * 4), \
                      (const void*)(Ab + (size_t)(n0 + srow) * NTOK + (m0_) + smo)); \
    }

    const int NS = NTOK / 32;                           // 128 k-steps
    AV_STAGE(0, 0)
    if (tid < 128) wait_asynccnt0();
    __syncthreads();

    for (int step = 0; step < NS; ++step) {
        const int cur = step & 1;
        const int m0  = step * 32;
        if (step + 1 < NS) { AV_STAGE(m0 + 32, cur ^ 1) }   // overlap with compute

        BF16Frag Bf;
        const float* ar = &atile[cur][l * 32 + half * 8];
#pragma unroll
        for (int j = 0; j < 4; ++j) {
            Bf.u[j]     = pack2bf(ar[2 * j],      ar[2 * j + 1]);
            Bf.u[j + 4] = pack2bf(ar[16 + 2 * j], ar[17 + 2 * j]);
        }
        BF16Frag A0, A1;
        const unsigned short* vr0 = Vb + (size_t)(c0 + l) * NTOK + m0 + half * 8;
        const unsigned short* vr1 = vr0 + (size_t)16 * NTOK;
#pragma unroll
        for (int j = 0; j < 4; ++j) {
            A0.u[j]     = ld_u32(vr0 + 2 * j);
            A0.u[j + 4] = ld_u32(vr0 + 16 + 2 * j);
            A1.u[j]     = ld_u32(vr1 + 2 * j);
            A1.u[j + 4] = ld_u32(vr1 + 16 + 2 * j);
        }
        acc0.v = wmma_bf16(A0, Bf, acc0);               // two independent chains
        acc1.v = wmma_bf16(A1, Bf, acc1);

        if (step + 1 < NS) {
            if (tid < 128) wait_asynccnt0();            // next buffer landed
            __syncthreads();                            // all readers done with cur
        }
    }
#undef AV_STAGE

    const float g = gamma[0];
#pragma unroll
    for (int r = 0; r < 8; ++r) {
        const int ncol = n0 + l;
        const int cr0  = c0 + r + 8 * half;
        const int cr1  = cr0 + 16;
        const size_t i0 = (size_t)b * CC * NTOK + (size_t)cr0 * NTOK + ncol;
        const size_t i1 = (size_t)b * CC * NTOK + (size_t)cr1 * NTOK + ncol;
        __builtin_nontemporal_store(g * acc0.f[r] + Xres[i0], &Out[i0]);
        __builtin_nontemporal_store(g * acc1.f[r] + Xres[i1], &Out[i1]);
    }
}

// ---------------------------------------------------------------------------
extern "C" void kernel_launch(void* const* d_in, const int* in_sizes, int n_in,
                              void* d_out, int out_size, void* d_ws, size_t ws_size,
                              hipStream_t stream)
{
    const float* x1     = (const float*)d_in[0];
    const float* x2     = (const float*)d_in[1];
    const float* x3     = (const float*)d_in[2];
    const float* xt     = (const float*)d_in[3];
    const float* Wq     = (const float*)d_in[4];
    const float* bq     = (const float*)d_in[5];
    const float* Wk     = (const float*)d_in[6];
    const float* bk     = (const float*)d_in[7];
    const float* Wk2    = (const float*)d_in[8];
    const float* bk2    = (const float*)d_in[9];
    const float* Wv     = (const float*)d_in[10];
    const float* bv     = (const float*)d_in[11];
    const float* Wv2    = (const float*)d_in[12];
    const float* bv2    = (const float*)d_in[13];
    const float* gamma  = (const float*)d_in[14];
    const float* gamma2 = (const float*)d_in[15];

    // Workspace layout (bf16 intermediates + f32 rowsums), ~23.1 MB
    unsigned short* qT  = (unsigned short*)d_ws;
    unsigned short* k1T = qT  + (size_t)BB * NTOK * CQ;
    unsigned short* k2T = k1T + (size_t)BB * NTOK * CQ;
    unsigned short* vW  = k2T + (size_t)BB * NTOK * CQ;
    unsigned short* vtW = vW  + (size_t)BB * CC * NTOK;
    float* rowsum = (float*)(vtW + (size_t)BB * CC * NTOK);

    float* attn = (float*)d_out;                         // (B,N,N)
    float* out1 = attn + (size_t)BB * NTOK * NTOK;       // (B,C,H,W)
    float* out2 = out1 + (size_t)BB * CC * NTOK;         // (B,C,H,W)

    const int qk_tiles = BB * (CQ / 16) * (NTOK / 16);   // 4096 tiles
    const int v_tiles  = BB * (CC / 16) * (NTOK / 16);   // 16384 tiles

    proj_kernel<1><<<qk_tiles / 8, 256, 0, stream>>>(Wq,  bq,  x3, qT,  CQ, CC);
    proj_kernel<1><<<qk_tiles / 8, 256, 0, stream>>>(Wk,  bk,  x1, k1T, CQ, CC);
    proj_kernel<1><<<qk_tiles / 8, 256, 0, stream>>>(Wk2, bk2, x2, k2T, CQ, CC);
    proj_kernel<0><<<v_tiles / 8,  256, 0, stream>>>(Wv,  bv,  x3, vW,  CC, CC);
    proj_kernel<0><<<v_tiles / 8,  256, 0, stream>>>(Wv2, bv2, xt, vtW, CC, CC);

    attn_kernel<<<BB * (NTOK / 16), 256, 0, stream>>>(qT, k1T, k2T, attn, rowsum);
    scale_kernel<<<BB * NTOK, 256, 0, stream>>>(attn, rowsum);

    av_kernel<<<BB * (NTOK / 16), 256, 0, stream>>>(vW,  attn, x3, gamma,  out1);
    av_kernel<<<BB * (NTOK / 16), 256, 0, stream>>>(vtW, attn, xt, gamma2, out2);
}
